// ConvSelfAttention_54485955117125
// MI455X (gfx1250) — matrix operand order repaired and checked by reading, not compiled, not run on hardware
//
#include <hip/hip_runtime.h>
#include <hip/hip_bf16.h>
#include <math.h>

// ---- problem constants ----
#define BB 4
#define CC 128
#define NN 4096
#define QBLKS (BB * NN / 16)   // 1024 query blocks of 16
#define RSQRT_C 0.08838834764831845f

typedef __attribute__((ext_vector_type(16))) _Float16 v16h;
typedef __attribute__((ext_vector_type(8)))  _Float16 v8h;
typedef __attribute__((ext_vector_type(8)))  float    v8f;
typedef __attribute__((ext_vector_type(4)))  float    v4f;

// D = A(16xK=32, f16) * B(32x16, f16) + C(16x16, f32)
__device__ __forceinline__ v8f wmma16(v16h a, v16h b, v8f c) {
  return __builtin_amdgcn_wmma_f32_16x16x32_f16(false, a, false, b, (short)0, c,
                                                false, false);
}

// A-tile (M=16 rows, K=32) from a row-major [row][k] f16 matrix, leading dim ld.
// A layout: lane L (m=L%16, h=L/16): VGPR0-3 <- k = h*16+0..7,
// VGPR4-7 <- k = h*16+8..15  => two contiguous 16B loads per lane.
__device__ __forceinline__ v16h load_A16(const _Float16* base, int ld, int row0,
                                         int k0, int lane) {
  const int m = row0 + (lane & 15);
  const int h = lane >> 4;
  const _Float16* p = base + (size_t)m * ld + k0 + h * 16;
  v8h lo = *(const v8h*)p;
  v8h hi = *(const v8h*)(p + 8);
  v16h a;
#pragma unroll
  for (int i = 0; i < 8; ++i) { a[i] = lo[i]; a[i + 8] = hi[i]; }
  return a;
}

// B-tile (K=32, N=16) where B(k,col) = src[col][k] (src row-major, ld = row len).
// 16-bit B layout used here (matches load_A16): lane L (col=L%16, h=L/16) holds k = h*16+0..15
// => one contiguous 32B load per lane.
__device__ __forceinline__ v16h load_B16(const _Float16* base, int ld, int col0,
                                         int k0, int lane) {
  const int c = col0 + (lane & 15);
  const int h = lane >> 4;
  return *(const v16h*)(base + (size_t)c * ld + k0 + h * 16);
}

// ---------------- setup: weight f32->f16, x transpose ----------------
__global__ void convert_weights(const float* wq, const float* wk, const float* wv,
                                const float* wo, _Float16* w16) {
  int i = blockIdx.x * blockDim.x + threadIdx.x;
  if (i < CC * CC) {
    w16[i]                 = (_Float16)wq[i];
    w16[CC * CC + i]       = (_Float16)wk[i];
    w16[2 * CC * CC + i]   = (_Float16)wv[i];
    w16[3 * CC * CC + i]   = (_Float16)wo[i];
  }
}

// x[b][c][n] f32 -> Xt[b][n][c] f16 (position-major for WMMA tile loads)
__global__ void transpose_x(const float* x, _Float16* Xt) {
  int i = blockIdx.x * blockDim.x + threadIdx.x;
  if (i < BB * CC * NN) {
    int n = i & (NN - 1);
    int c = (i >> 12) & (CC - 1);
    int b = i >> 19;
    Xt[((size_t)b * NN + n) * CC + c] = (_Float16)x[i];
  }
}

// ---------------- QKV projection ----------------
// Qt/Kt: [B][N][C] f16 (Q pre-scaled by 1/sqrt(C)); Vc: [B][C][N] f16
__global__ void __launch_bounds__(128) qkv_kernel(
    const _Float16* __restrict__ Xt, const _Float16* __restrict__ w16,
    const float* __restrict__ bq, const float* __restrict__ bk,
    const float* __restrict__ bv, _Float16* __restrict__ Qt,
    _Float16* __restrict__ Kt, _Float16* __restrict__ Vc) {
  const int lane = threadIdx.x & 31;
  const int qblk = blockIdx.x * 4 + (threadIdx.x >> 5);
  const int b  = qblk >> 8;              // 256 n-blocks per batch
  const int n0 = (qblk & 255) << 4;
  const int col = lane & 15, half = lane >> 4;

  const _Float16* Xb   = Xt + (size_t)b * NN * CC;
  const _Float16* wq16 = w16;
  const _Float16* wk16 = w16 + CC * CC;
  const _Float16* wv16 = w16 + 2 * CC * CC;

  v16h xB[4], xA[4];
#pragma unroll
  for (int kc = 0; kc < 4; ++kc) {
    xB[kc] = load_B16(Xb, CC, n0, kc * 32, lane);  // B(k=c, col=n)
    xA[kc] = load_A16(Xb, CC, n0, kc * 32, lane);  // A(m=n, k=c)
  }

  const v8f z = {0.f, 0.f, 0.f, 0.f, 0.f, 0.f, 0.f, 0.f};
#pragma unroll
  for (int ot = 0; ot < 8; ++ot) {
    // hoist all weight tiles for this output tile so loads clause together
    v16h wqA[4], wkA[4], wvB[4];
#pragma unroll
    for (int kc = 0; kc < 4; ++kc) {
      wqA[kc] = load_A16(wq16, CC, ot * 16, kc * 32, lane);
      wkA[kc] = load_A16(wk16, CC, ot * 16, kc * 32, lane);
      wvB[kc] = load_B16(wv16, CC, ot * 16, kc * 32, lane);
    }
    // Q,K: D[o,n] = sum_c w[o,c] x[c,n]   (rows=o, cols=n)
    v8f dq = z, dk = z, dv = z;
#pragma unroll
    for (int kc = 0; kc < 4; ++kc) {
      dq = wmma16(wqA[kc], xB[kc], dq);
      dk = wmma16(wkA[kc], xB[kc], dk);
      dv = wmma16(xA[kc], wvB[kc], dv);   // V: D[n,o] (rows=n, cols=o)
    }
    v8h oq, ok;
#pragma unroll
    for (int v = 0; v < 8; ++v) {
      int o = ot * 16 + half * 8 + v;    // D row o = v + 8*half
      oq[v] = (_Float16)((dq[v] + bq[o]) * RSQRT_C);
      ok[v] = (_Float16)(dk[v] + bk[o]);
    }
    size_t off = ((size_t)b * NN + n0 + col) * CC + ot * 16 + half * 8;
    *(v8h*)(Qt + off) = oq;
    *(v8h*)(Kt + off) = ok;

    float bias = bv[ot * 16 + col];
    v8h ov;
#pragma unroll
    for (int v = 0; v < 8; ++v) ov[v] = (_Float16)(dv[v] + bias);
    *(v8h*)(Vc + ((size_t)b * CC + ot * 16 + col) * NN + n0 + half * 8) = ov;
  }
}

// ---------------- flash attention ----------------
// One pipeline stage: consume kCur (register-resident K tiles for chunk m0),
// while issuing loads of V tiles for m0 and K tiles for chunk mn into kNext.
__device__ __forceinline__ void attn_chunk(
    const _Float16* __restrict__ Kb, const _Float16* __restrict__ Vb,
    int lane, bool hiHalf, int m0, int mn, const v16h qB[4],
    v16h kCur[8], v16h kNext[8], v8f acc[8], float& m_i, float& l_i) {
  const v8f z = {0.f, 0.f, 0.f, 0.f, 0.f, 0.f, 0.f, 0.f};

  // issue loads first: latency hides under S-WMMAs + softmax VALU stretch
  v16h vA[8];
#pragma unroll
  for (int ct = 0; ct < 8; ++ct)
    vA[ct] = load_A16(Vb, NN, ct * 16, m0, lane);   // A(rows=c, k=m)
#pragma unroll
  for (int kc = 0; kc < 4; ++kc) {
    kNext[kc * 2]     = load_A16(Kb, CC, mn,      kc * 32, lane);
    kNext[kc * 2 + 1] = load_A16(Kb, CC, mn + 16, kc * 32, lane);
  }

  // S^T[m,n] = sum_c K^T[m,c] * Q^T[n,c]  (Q already holds 1/sqrt(C));
  // kCur tiles already in registers -> no load wait.
  v8f s1 = z, s2 = z;
#pragma unroll
  for (int kc = 0; kc < 4; ++kc) {
    s1 = wmma16(kCur[kc * 2],     qB[kc], s1);
    s2 = wmma16(kCur[kc * 2 + 1], qB[kc], s2);
  }

  // online softmax over keys for this lane's query column
  float mx = m_i;
#pragma unroll
  for (int v = 0; v < 8; ++v) mx = fmaxf(mx, fmaxf(s1[v], s2[v]));
  mx = fmaxf(mx, __shfl_xor(mx, 16, 32));
  float alpha = __expf(m_i - mx);
  float p1[8], p2[8], ls = 0.f;
#pragma unroll
  for (int v = 0; v < 8; ++v) {
    p1[v] = __expf(s1[v] - mx);
    p2[v] = __expf(s2[v] - mx);
    ls += p1[v] + p2[v];
  }
  ls += __shfl_xor(ls, 16, 32);
  l_i = l_i * alpha + ls;
  m_i = mx;

  // Build P^T as B operand (k=key 0..31 in lane half, col=query):
  // lanes 0-15 need all 16 rows of S1; lanes 16-31 all 16 rows of S2.
  v16h pB;
#pragma unroll
  for (int v = 0; v < 8; ++v) {
    float o1 = __shfl_xor(p1[v], 16, 32);
    float o2 = __shfl_xor(p2[v], 16, 32);
    pB[v]     = (_Float16)(hiHalf ? o2    : p1[v]);   // tile row v
    pB[v + 8] = (_Float16)(hiHalf ? p2[v] : o1);      // tile row v+8
  }

  // O^T[c,n] = alpha*O^T + sum_m V[c,m] * P^T[m,n]
#pragma unroll
  for (int ct = 0; ct < 8; ++ct) {
#pragma unroll
    for (int v = 0; v < 8; ++v) acc[ct][v] *= alpha;
    acc[ct] = wmma16(vA[ct], pB, acc[ct]);
  }
}

// One wave handles 16 queries; keys streamed 32 at a time, pipelined 2-deep
// with ping/pong K register buffers (no rotation movs).
__global__ void __launch_bounds__(128) flash_kernel(
    const _Float16* __restrict__ Qt, const _Float16* __restrict__ Kt,
    const _Float16* __restrict__ Vc, _Float16* __restrict__ Ot) {
  const int lane = threadIdx.x & 31;
  const int qblk = blockIdx.x * 4 + (threadIdx.x >> 5);
  const int b  = qblk >> 8;
  const int n0 = (qblk & 255) << 4;
  const bool hiHalf = lane >= 16;

  const _Float16* Qb = Qt + (size_t)b * NN * CC;
  const _Float16* Kb = Kt + (size_t)b * NN * CC;
  const _Float16* Vb = Vc + (size_t)b * CC * NN;

  v16h qB[4];                               // resident Q B-tiles (k=c, col=n)
#pragma unroll
  for (int kc = 0; kc < 4; ++kc) qB[kc] = load_B16(Qb, CC, n0, kc * 32, lane);

  const v8f z = {0.f, 0.f, 0.f, 0.f, 0.f, 0.f, 0.f, 0.f};
  v8f acc[8];                               // O^T tiles: rows=c, cols=n
#pragma unroll
  for (int ct = 0; ct < 8; ++ct) acc[ct] = z;

  float m_i = -INFINITY, l_i = 0.f;

  // prologue: K tiles for chunk 0 into ping buffer
  v16h kA[8], kB[8];
#pragma unroll
  for (int kc = 0; kc < 4; ++kc) {
    kA[kc * 2]     = load_A16(Kb, CC, 0,  kc * 32, lane);
    kA[kc * 2 + 1] = load_A16(Kb, CC, 16, kc * 32, lane);
  }

  for (int m0 = 0; m0 < NN; m0 += 64) {
    // stage 1: consume kA, prefetch K(m0+32) into kB
    attn_chunk(Kb, Vb, lane, hiHalf, m0, m0 + 32, qB, kA, kB, acc, m_i, l_i);
    // stage 2: consume kB, prefetch K(m0+64) into kA (last iter: dummy reload)
    const int mn = (m0 + 64 < NN) ? (m0 + 64) : 0;
    attn_chunk(Kb, Vb, lane, hiHalf, m0 + 32, mn, qB, kB, kA, acc, m_i, l_i);
  }

  const float inv_l = 1.f / l_i;
  const int n = n0 + (lane & 15), half = lane >> 4;
#pragma unroll
  for (int ct = 0; ct < 8; ++ct) {
    v8h o;
#pragma unroll
    for (int v = 0; v < 8; ++v) o[v] = (_Float16)(acc[ct][v] * inv_l);
    *(v8h*)(Ot + ((size_t)b * NN + n) * CC + ct * 16 + half * 8) = o;
  }
}

// ---------------- output projection + residual ----------------
// y[o,n] = gamma*(sum_c wo[o,c] O[c,n] + bo[o]) + x[o,n]
__global__ void __launch_bounds__(128) proj_out_kernel(
    const _Float16* __restrict__ Ot, const _Float16* __restrict__ wo16,
    const float* __restrict__ bo, const float* __restrict__ gamma,
    const float* __restrict__ x, float* __restrict__ y) {
  const int lane = threadIdx.x & 31;
  const int qblk = blockIdx.x * 4 + (threadIdx.x >> 5);
  const int b  = qblk >> 8;
  const int n0 = (qblk & 255) << 4;
  const int col = lane & 15, half = lane >> 4;
  const float g = gamma[0];

  const _Float16* Ob = Ot + (size_t)b * NN * CC;
  v16h oA[4];
#pragma unroll
  for (int kc = 0; kc < 4; ++kc) oA[kc] = load_A16(Ob, CC, n0, kc * 32, lane);

  const v8f z = {0.f, 0.f, 0.f, 0.f, 0.f, 0.f, 0.f, 0.f};
#pragma unroll
  for (int ot = 0; ot < 8; ++ot) {
    v16h woB[4];
#pragma unroll
    for (int kc = 0; kc < 4; ++kc)
      woB[kc] = load_B16(wo16, CC, ot * 16, kc * 32, lane);
    v8f d = z;
#pragma unroll
    for (int kc = 0; kc < 4; ++kc) d = wmma16(oA[kc], woB[kc], d);
    int o = ot * 16 + col;                 // D cols = o, rows = n (contiguous)
    float bias = bo[o];
    size_t base = ((size_t)b * CC + o) * NN + n0 + half * 8;
    v4f x0 = *(const v4f*)(x + base);
    v4f x1 = *(const v4f*)(x + base + 4);
    v4f y0, y1;
#pragma unroll
    for (int i = 0; i < 4; ++i) {
      y0[i] = g * (d[i] + bias) + x0[i];
      y1[i] = g * (d[i + 4] + bias) + x1[i];
    }
    *(v4f*)(y + base) = y0;
    *(v4f*)(y + base + 4) = y1;
  }
}

extern "C" void kernel_launch(void* const* d_in, const int* in_sizes, int n_in,
                              void* d_out, int out_size, void* d_ws, size_t ws_size,
                              hipStream_t stream) {
  const float* x  = (const float*)d_in[0];
  const float* wq = (const float*)d_in[1];
  const float* bq = (const float*)d_in[2];
  const float* wk = (const float*)d_in[3];
  const float* bk = (const float*)d_in[4];
  const float* wv = (const float*)d_in[5];
  const float* bv = (const float*)d_in[6];
  const float* wo = (const float*)d_in[7];
  const float* bo = (const float*)d_in[8];
  const float* gm = (const float*)d_in[9];
  float* y = (float*)d_out;

  char* ws = (char*)d_ws;
  const size_t TEN = (size_t)BB * NN * CC * sizeof(_Float16);  // 4 MiB
  _Float16* Xt  = (_Float16*)(ws + 0 * TEN);  // reused as Ot after QKV
  _Float16* Qt  = (_Float16*)(ws + 1 * TEN);
  _Float16* Kt  = (_Float16*)(ws + 2 * TEN);
  _Float16* Vc  = (_Float16*)(ws + 3 * TEN);
  _Float16* w16 = (_Float16*)(ws + 4 * TEN);  // 4 * C*C halves = 128 KiB
  _Float16* Ot  = Xt;

  convert_weights<<<(CC * CC + 255) / 256, 256, 0, stream>>>(wq, wk, wv, wo, w16);
  transpose_x<<<(BB * CC * NN + 255) / 256, 256, 0, stream>>>(x, Xt);
  qkv_kernel<<<QBLKS / 4, 128, 0, stream>>>(Xt, w16, bq, bk, bv, Qt, Kt, Vc);
  flash_kernel<<<QBLKS / 4, 128, 0, stream>>>(Qt, Kt, Vc, Ot);
  proj_out_kernel<<<QBLKS / 4, 128, 0, stream>>>(Ot, w16 + 3 * CC * CC, bo, gm, x, y);
}
